// _SpatialAttention_35940286332967
// MI455X (gfx1250) — matrix operand order repaired and checked by reading, not compiled, not run on hardware
//
#include <hip/hip_runtime.h>
#include <hip/hip_bf16.h>

// ---------------------------------------------------------------------------
// Problem constants (from reference):
//   B=16, N=4096, C=4, P=64 -> CP=256 features, D=128, WIN=7, ALPHA=0.5
//   Rows per matrix: B*N = 65536.  Two matrices (x, x_template) -> 131072 rows.
// ---------------------------------------------------------------------------
#define NB        16
#define NSEQ      4096
#define CP        256          // C*P
#define DDIM      128
#define ROWS      65536        // B*N
#define ROWS2     131072       // both matrices
#define TILES     8192         // ROWS2 / 16
#define TILES_MAT 4096
#define GEMM_BLOCKS 256        // 8 waves/block * 4 iters = 8192 tiles
#define EPS_F     1e-5f

typedef __attribute__((ext_vector_type(16))) __bf16 bf16x16;
typedef __attribute__((ext_vector_type(8)))  __bf16 bf16x8;
typedef __attribute__((ext_vector_type(8)))  float  f32x8;

// f32 -> bf16 (RNE). Plain fptrunc lets the gfx1250 backend pick its native
// conversion ops instead of a 3-VALU-op bit twiddle per element.
__device__ __forceinline__ __bf16 f2bf(float f) {
    return static_cast<__bf16>(f);
}

// XOR-swizzled LDS index for W (bf16), 16B-chunk granularity -> conflict-free
// B-fragment reads (lanes 0..15 hit distinct chunk slots). 128*256*2B = 64KB.
__device__ __forceinline__ int wl_idx(int d, int k) {
    int chunk = (k >> 3) ^ (d & 15);          // 32 chunks of 8 halves per row
    return d * 256 + ((chunk & 31) << 3) + (k & 7);
}

// ---------------------------------------------------------------------------
// Kernel 1: embedding GEMM  h[r][d] = sum_k X[r][k] * W[d][k]
//   bf16 WMMA (16x16x32), f32 accumulate. Bias b omitted: it cancels exactly
//   in the subsequent mean-subtraction.
//   Also emits deterministic per-tile column partial sums / sums-of-squares.
// ---------------------------------------------------------------------------
__global__ __launch_bounds__(256)
void k_embed_gemm(const float* __restrict__ x, const float* __restrict__ xt,
                  const float* __restrict__ W, float* __restrict__ h,
                  float* __restrict__ psum, float* __restrict__ psq)
{
    extern __shared__ __bf16 Wl[];            // 128 x 256 bf16, swizzled
    const int tid = threadIdx.x;

    // Stage W -> LDS (bf16), coalesced global reads.
    for (int i = tid; i < DDIM * CP; i += 256) {
        int d = i >> 8, k = i & 255;
        Wl[wl_idx(d, k)] = f2bf(W[i]);
    }
    __syncthreads();

    const int wave  = tid >> 5;
    const int lane  = tid & 31;
    const int lhalf = lane >> 4;              // 0 | 1
    const int lmod  = lane & 15;

    for (int iter = 0; iter < 4; ++iter) {
        const int tile = blockIdx.x * 8 + wave + iter * 2048;   // 0..8191
        const float* src;
        int row0;
        if (tile < TILES_MAT) { src = x;  row0 = tile * 16; }
        else                  { src = xt; row0 = (tile - TILES_MAT) * 16; }
        const float* arow = src + (size_t)(row0 + lmod) * CP;   // lane's A row

        f32x8 acc[8];
        #pragma unroll
        for (int t = 0; t < 8; ++t) acc[t] = (f32x8)0.0f;

        #pragma unroll
        for (int ks = 0; ks < 8; ++ks) {      // K = 256 in steps of 32
            // A fragment: lane (m = lmod) holds k = kb..kb+7 and kb+16..kb+23
            const int kb = ks * 32 + lhalf * 8;
            float4 a0 = *(const float4*)(arow + kb);
            float4 a1 = *(const float4*)(arow + kb + 4);
            float4 a2 = *(const float4*)(arow + kb + 16);
            float4 a3 = *(const float4*)(arow + kb + 20);
            bf16x16 af;
            af[0] = f2bf(a0.x);  af[1] = f2bf(a0.y);  af[2]  = f2bf(a0.z);  af[3]  = f2bf(a0.w);
            af[4] = f2bf(a1.x);  af[5] = f2bf(a1.y);  af[6]  = f2bf(a1.z);  af[7]  = f2bf(a1.w);
            af[8] = f2bf(a2.x);  af[9] = f2bf(a2.y);  af[10] = f2bf(a2.z);  af[11] = f2bf(a2.w);
            af[12] = f2bf(a3.x); af[13] = f2bf(a3.y); af[14] = f2bf(a3.z);  af[15] = f2bf(a3.w);

            const int krow = ks * 32 + lhalf * 8;    // B: lane holds K=krow..krow+7 and krow+16..krow+23
            #pragma unroll
            for (int dt = 0; dt < 8; ++dt) {
                const int d = dt * 16 + lmod;
                bf16x8 blo = *(const bf16x8*)&Wl[wl_idx(d, krow)];
                bf16x8 bhi = *(const bf16x8*)&Wl[wl_idx(d, krow + 16)];
                bf16x16 bf = __builtin_shufflevector(blo, bhi,
                    0, 1, 2, 3, 4, 5, 6, 7, 8, 9, 10, 11, 12, 13, 14, 15);
                acc[dt] = __builtin_amdgcn_wmma_f32_16x16x32_bf16(
                    false, af, false, bf, (short)0, acc[dt], false, false);
            }
        }

        // Epilogue: write h tile (f32) + deterministic column partials.
        float* hbase = h + (size_t)tile * 16 * DDIM;
        #pragma unroll
        for (int dt = 0; dt < 8; ++dt) {
            float cs = 0.f, cq = 0.f;
            #pragma unroll
            for (int v = 0; v < 8; ++v) {      // C/D layout: m = v + 8*lhalf
                const int m = v + lhalf * 8;
                float val = acc[dt][v];
                hbase[(size_t)m * DDIM + dt * 16 + lmod] = val;
                cs += val;
                cq += val * val;
            }
            cs += __shfl_xor(cs, 16);          // fold the two 8-row halves
            cq += __shfl_xor(cq, 16);
            if (lhalf == 0) {
                psum[(size_t)tile * DDIM + dt * 16 + lmod] = cs;
                psq [(size_t)tile * DDIM + dt * 16 + lmod] = cq;
            }
        }
    }
}

// ---------------------------------------------------------------------------
// Kernel 2: finalize stats -> per-column scale/shift. grid(2), block(128).
//   ss layout: [mat*256 + d] = scale, [mat*256 + 128 + d] = shift
// ---------------------------------------------------------------------------
__global__ void k_stats(const float* __restrict__ psum, const float* __restrict__ psq,
                        const float* __restrict__ gamma, const float* __restrict__ beta,
                        float* __restrict__ ss)
{
    const int mat = blockIdx.x;
    const int d   = threadIdx.x;
    const int base = mat * TILES_MAT;
    float s = 0.f, q = 0.f;
    for (int t = 0; t < TILES_MAT; ++t) {      // fixed order -> deterministic
        s += psum[(size_t)(base + t) * DDIM + d];
        q += psq [(size_t)(base + t) * DDIM + d];
    }
    const float mean = s * (1.0f / (float)ROWS);
    const float var  = q * (1.0f / (float)ROWS) - mean * mean;
    const float rstd = 1.0f / sqrtf(var + EPS_F);
    const float sc = gamma[d] * rstd;
    ss[mat * 256 + d]       = sc;
    ss[mat * 256 + 128 + d] = beta[d] - mean * sc;
}

// ---------------------------------------------------------------------------
// Kernel 3: windowed attention. One wave per position (65536 total).
//   Normalization + leaky-ReLU fused into the h reads.
// ---------------------------------------------------------------------------
__global__ __launch_bounds__(256)
void k_attn(const float* __restrict__ x, const float* __restrict__ xt,
            const float* __restrict__ h, const float* __restrict__ ss,
            float* __restrict__ out, float* __restrict__ sim_out)
{
    const int wave = threadIdx.x >> 5;
    const int lane = threadIdx.x & 31;
    const size_t nf = (size_t)blockIdx.x * 8 + wave;   // flat (b,n), 0..65535
    const int bi = (int)(nf >> 12);
    const int n  = (int)(nf & 4095);

    const float* h_t = h + (size_t)ROWS * DDIM;

    // Window columns (clamped) + early prefetch of the xt rows that are only
    // consumed after sim/softmax: lanes 32B apart cover each 1KB row.
    int cols[7];
    #pragma unroll
    for (int w = 0; w < 7; ++w) {
        int c = n - 3 + w;
        c = c < 0 ? 0 : (c > NSEQ - 1 ? NSEQ - 1 : c);
        cols[w] = c;
        __builtin_prefetch(xt + ((size_t)bi * NSEQ + c) * CP + lane * 8, 0, 3);
    }

    // normalized+activated query embedding, 4 d-components per lane
    float sct[4], sht[4], ex[4];
    #pragma unroll
    for (int k = 0; k < 4; ++k) {
        const int d = lane + 32 * k;
        const float scx = ss[d], shx = ss[128 + d];
        sct[k] = ss[256 + d];
        sht[k] = ss[384 + d];
        float v = h[nf * DDIM + d] * scx + shx;
        ex[k] = v >= 0.f ? v : 0.1f * v;
    }

    float s[7];
    #pragma unroll
    for (int w = 0; w < 7; ++w) {
        const float* ht = h_t + ((size_t)bi * NSEQ + cols[w]) * DDIM;
        float a = 0.f;
        #pragma unroll
        for (int k = 0; k < 4; ++k) {
            const int d = lane + 32 * k;
            float v = ht[d] * sct[k] + sht[k];
            v = v >= 0.f ? v : 0.1f * v;
            a += ex[k] * v;
        }
        #pragma unroll
        for (int off = 16; off > 0; off >>= 1) a += __shfl_xor(a, off);
        s[w] = a;
    }

    if (lane == 0) {
        #pragma unroll
        for (int w = 0; w < 7; ++w) sim_out[nf * 7 + w] = s[w];
    }

    // masked softmax over the window (center always valid)
    float mx = -3.4e38f;
    #pragma unroll
    for (int w = 0; w < 7; ++w) {
        const bool valid = (n - 3 + w >= 0) && (n - 3 + w < NSEQ);
        if (valid && s[w] > mx) mx = s[w];
    }
    float e[7], esum = 0.f;
    #pragma unroll
    for (int w = 0; w < 7; ++w) {
        const bool valid = (n - 3 + w >= 0) && (n - 3 + w < NSEQ);
        e[w] = valid ? __expf(s[w] - mx) : 0.f;
        esum += e[w];
    }
    const float inv = 1.0f / esum;

    // blended output: 0.5*x + 0.5*attn-weighted template window
    const float* xrow = x + nf * CP;
    float* orow = out + nf * CP;
    #pragma unroll
    for (int k2 = 0; k2 < 8; ++k2) {
        const int ep = lane + 32 * k2;
        float a = 0.f;
        #pragma unroll
        for (int w = 0; w < 7; ++w)
            a += e[w] * xt[((size_t)bi * NSEQ + cols[w]) * CP + ep];
        orow[ep] = 0.5f * xrow[ep] + 0.5f * (a * inv);
    }
}

// ---------------------------------------------------------------------------
extern "C" void kernel_launch(void* const* d_in, const int* in_sizes, int n_in,
                              void* d_out, int out_size, void* d_ws, size_t ws_size,
                              hipStream_t stream)
{
    const float* x     = (const float*)d_in[0];
    const float* xtem  = (const float*)d_in[1];
    const float* W     = (const float*)d_in[2];
    // d_in[3] = b: mathematically cancels in the mean subtraction -> unused.
    const float* gamma = (const float*)d_in[4];
    const float* beta  = (const float*)d_in[5];

    float* out = (float*)d_out;                          // (B,N,C,P) = 16.7M f32
    float* sim = out + (size_t)ROWS * CP;                // (B,N,7)

    // workspace: h (131072x128 f32 = 64MB) + partials (2x 8192x128 f32) + ss
    float* h    = (float*)d_ws;
    float* psum = h    + (size_t)ROWS2 * DDIM;
    float* psq  = psum + (size_t)TILES * DDIM;
    float* ss   = psq  + (size_t)TILES * DDIM;

    const size_t shmem = (size_t)DDIM * CP * sizeof(__bf16);   // 64 KB
    k_embed_gemm<<<GEMM_BLOCKS, 256, shmem, stream>>>(x, xtem, W, h, psum, psq);
    k_stats<<<2, 128, 0, stream>>>(psum, psq, gamma, beta, ss);
    k_attn<<<ROWS / 8, 256, 0, stream>>>(x, xtem, h, ss, out, sim);
}
